// EncoderWithGSMK_14448269984292
// MI455X (gfx1250) — compile-verified, matrix operand order checked by reading, not run on hardware
//
#include <hip/hip_runtime.h>
#include <hip/hip_bf16.h>

#define NL 2
#define DMODEL 512
#define HHEADS 8
#define DH 64
#define DFF 2048
#define TD 8
#define BB 8
#define LL 512
#define MROWS (BB * LL) // 4096

typedef __attribute__((ext_vector_type(16))) __bf16 v16bf;
typedef __attribute__((ext_vector_type(8)))  __bf16 v8bf;
typedef __attribute__((ext_vector_type(8)))  float  v8f;
typedef int v4i_vs __attribute__((vector_size(16)));

#define AS1 __attribute__((address_space(1)))
#define AS3 __attribute__((address_space(3)))

#if defined(__has_builtin)
#if __has_builtin(__builtin_amdgcn_global_load_async_to_lds_b128) && \
    __has_builtin(__builtin_amdgcn_s_wait_asynccnt)
#define HAVE_ASYNC 1
#endif
#endif
#ifndef HAVE_ASYNC
#define HAVE_ASYNC 0
#endif

// ---- WMMA 16x16x32 bf16 element mapping (ISA 7.12.2 16-bit layouts) ----
__device__ __forceinline__ int kmap(int e, int half) {
  return (e < 8) ? (half * 8 + e) : (16 + half * 8 + (e - 8));
}

// A: 16x32 tile from row-major activations (e0..7 / e8..15 are contiguous -> 2x b128)
__device__ __forceinline__ v16bf frag_a(const __bf16* A, int lda, int m0, int k0, int lane) {
  v16bf f;
  int half = lane >> 4, r = lane & 15;
  const __bf16* p = A + (size_t)(m0 + r) * lda + k0;
#pragma unroll
  for (int e = 0; e < 16; ++e) f[e] = p[kmap(e, half)];
  return f;
}

// ---- fp32 -> bf16 convert ----
__global__ void f2bf_kernel(const float* __restrict__ s, __bf16* __restrict__ d, int n) {
  int i = blockIdx.x * blockDim.x + threadIdx.x;
  if (i < n) d[i] = (__bf16)s[i];
}

// ---- pack fp32 weight (KxN row-major) into WMMA B-fragment order:
// P[((kt*(N/16)+nt)*32 + lane)*16 + e] = W[(kt*32 + kmap(e,half))*N + nt*16 + c]
__global__ void pack_w_kernel(const float* __restrict__ W, __bf16* __restrict__ P, int K, int N) {
  int idx = blockIdx.x * blockDim.x + threadIdx.x;
  if (idx >= K * N) return;
  int e = idx & 15, lane = (idx >> 4) & 31, tile = idx >> 9;
  int ntn = N >> 4;
  int nt = tile % ntn, kt = tile / ntn;
  int half = lane >> 4, c = lane & 15;
  P[idx] = (__bf16)W[(size_t)(kt * 32 + kmap(e, half)) * N + nt * 16 + c];
}

// ---- pack K (row-major bf16 [B*L, D]) into K^T B-fragments per (b,h):
// index = ((bh*2 + kt2)*32 + nt)*512 + lane*16 + e
__global__ void pack_k_kernel(const __bf16* __restrict__ kbuf, __bf16* __restrict__ P) {
  int idx = blockIdx.x * blockDim.x + threadIdx.x;
  if (idx >= BB * HHEADS * 2 * (LL / 16) * 512) return;
  int e = idx & 15, lane = (idx >> 4) & 31;
  int nt = (idx >> 9) & 31;
  int kt2 = (idx >> 14) & 1;
  int bh = idx >> 15;
  int b = bh >> 3, h = bh & 7;
  int half = lane >> 4, c = lane & 15;
  int kd = kt2 * 32 + kmap(e, half);
  int key = nt * 16 + c;
  P[idx] = kbuf[((size_t)b * LL + key) * DMODEL + h * DH + kd];
}

// ---- pack V into B-fragments per (b,h): index = ((bh*16 + ktile)*4 + nt)*512 + lane*16 + e
__global__ void pack_v_kernel(const __bf16* __restrict__ vbuf, __bf16* __restrict__ P) {
  int idx = blockIdx.x * blockDim.x + threadIdx.x;
  if (idx >= BB * HHEADS * (LL / 32) * 4 * 512) return;
  int e = idx & 15, lane = (idx >> 4) & 31;
  int nt = (idx >> 9) & 3;
  int ktile = (idx >> 11) & 15;
  int bh = idx >> 15;
  int b = bh >> 3, h = bh & 7;
  int half = lane >> 4, c = lane & 15;
  int kk = ktile * 32 + kmap(e, half);
  P[idx] = vbuf[((size_t)b * LL + kk) * DMODEL + h * DH + nt * 16 + c];
}

// ---- stage one 16x32 bf16 A tile into LDS (async on CDNA5, 64 lanes x b128) ----
__device__ __forceinline__ void stage_a(const __bf16* A, int m0, int k0, int K,
                                        __bf16* dst, int tid) {
  if (tid < 64) {
    int row = tid >> 2, seg = tid & 3;
    const __bf16* g = A + (size_t)(m0 + row) * K + k0 + seg * 8;
    __bf16* l = dst + row * 32 + seg * 8;
#if HAVE_ASYNC
    __builtin_amdgcn_global_load_async_to_lds_b128((AS1 v4i_vs*)g, (AS3 v4i_vs*)l, 0, 0);
#else
    *(v8bf*)l = *(const v8bf*)g;
#endif
  }
}

// ---- generic WMMA GEMM: C[M,N] = A[M,K] * B + bias (B pre-packed), opt relu ----
// 4 waves/block: one 16x16 tile each across 64 columns; A double-buffered in LDS
__global__ __launch_bounds__(128) void gemm_bf16(
    const __bf16* __restrict__ A, const __bf16* __restrict__ Bp,
    const float* __restrict__ bias, float* Cf, __bf16* Cb,
    int M, int N, int K, int relu) {
  __shared__ __bf16 As[2][16 * 32];
  int tid = threadIdx.x;
  int lane = tid & 31, wave = tid >> 5;
  int m0 = blockIdx.x * 16;
  int n0 = blockIdx.y * 64 + wave * 16;
  int ntn = N >> 4;
  int nsteps = K >> 5;
  int r = lane & 15, half = lane >> 4;

  stage_a(A, m0, 0, K, As[0], tid);
  v8f acc = {};
  for (int s = 0; s < nsteps; ++s) {
    int cur = s & 1;
#if HAVE_ASYNC
    __builtin_amdgcn_s_wait_asynccnt(0);
#endif
    __syncthreads();
    if (s + 1 < nsteps) stage_a(A, m0, (s + 1) << 5, K, As[cur ^ 1], tid);
    const __bf16* bptr = Bp + ((size_t)s * ntn + (n0 >> 4)) * 512 + lane * 16;
    if (s + 1 < nsteps)
      __builtin_prefetch(Bp + ((size_t)(s + 1) * ntn + (n0 >> 4)) * 512 + lane * 16, 0, 1);
    v8bf lo = *(const v8bf*)&As[cur][r * 32 + half * 8];
    v8bf hi = *(const v8bf*)&As[cur][r * 32 + 16 + half * 8];
    v16bf a = __builtin_shufflevector(lo, hi, 0, 1, 2, 3, 4, 5, 6, 7, 8, 9, 10, 11, 12, 13, 14, 15);
    v16bf b = *(const v16bf*)bptr;
    acc = __builtin_amdgcn_wmma_f32_16x16x32_bf16(false, a, false, b, (short)0, acc, false, false);
  }
  int col = n0 + (lane & 15);
  float bv = bias ? bias[col] : 0.0f;
#pragma unroll
  for (int i = 0; i < 8; ++i) {
    int row = m0 + i + half * 8;
    float v = acc[i] + bv;
    if (relu) v = fmaxf(v, 0.0f);
    size_t idx = (size_t)row * N + col;
    if (Cf) Cf[idx] = v;
    if (Cb) Cb[idx] = (__bf16)v;
  }
}

// ---- tiny TD=8 time FFNs: pv->tp, sg, be ----
__device__ __forceinline__ void ffn2_dev(float tv, const float* w1, const float* b1,
                                         const float* w2, const float* b2, float* out) {
  float h[TD];
#pragma unroll
  for (int j = 0; j < TD; ++j) h[j] = fmaxf(tv * w1[j] + b1[j], 0.0f);
#pragma unroll
  for (int j = 0; j < TD; ++j) {
    float s = b2[j];
#pragma unroll
    for (int i = 0; i < TD; ++i) s += h[i] * w2[i * TD + j];
    out[j] = fmaxf(s, 0.0f);
  }
}

__global__ void time_ffn_kernel(
    const float* __restrict__ t,
    const float* __restrict__ pw1, const float* __restrict__ pb1,
    const float* __restrict__ pw2, const float* __restrict__ pb2,
    const float* __restrict__ sw1, const float* __restrict__ sb1,
    const float* __restrict__ sw2, const float* __restrict__ sb2,
    const float* __restrict__ bw1, const float* __restrict__ bb1,
    const float* __restrict__ bw2, const float* __restrict__ bb2,
    float* __restrict__ tp, float* __restrict__ sg, float* __restrict__ be) {
  int idx = blockIdx.x * blockDim.x + threadIdx.x;
  if (idx >= BB * LL) return;
  float tv = t[idx];
  float o[TD];
  ffn2_dev(tv, pw1, pb1, pw2, pb2, o);
#pragma unroll
  for (int j = 0; j < TD; ++j) tp[idx * TD + j] = 6.28318530717958647692f * o[j] * tv;
  ffn2_dev(tv, sw1, sb1, sw2, sb2, o);
#pragma unroll
  for (int j = 0; j < TD; ++j) sg[idx * TD + j] = o[j] + 1e-6f;
  ffn2_dev(tv, bw1, bb1, bw2, bb2, o);
#pragma unroll
  for (int j = 0; j < TD; ++j) be[idx * TD + j] = o[j];
}

// ---- kernel bias: kern[b,q,k] * exp(-sqrt(dh)) ----
__global__ void kern_kernel(const float* __restrict__ t, const float* __restrict__ tp,
                            const float* __restrict__ sg, const float* __restrict__ be,
                            float* __restrict__ out) {
  int idx = blockIdx.x * blockDim.x + threadIdx.x;
  if (idx >= BB * LL * LL) return;
  int b = idx / (LL * LL);
  int rem = idx - b * (LL * LL);
  int q = rem / LL, k = rem - (rem / LL) * LL;
  const float* tq = tp + ((size_t)b * LL + q) * TD;
  const float* tk = tp + ((size_t)b * LL + k) * TD;
  const float* sq = sg + ((size_t)b * LL + q) * TD;
  const float* sk = sg + ((size_t)b * LL + k) * TD;
  const float* eq = be + ((size_t)b * LL + q) * TD;
  const float* ek = be + ((size_t)b * LL + k) * TD;
  float dt = t[b * LL + q] - t[b * LL + k];
  float d2 = dt * dt;
  float sum = 0.0f;
#pragma unroll
  for (int d = 0; d < TD; ++d) {
    float a = sq[d], c = sk[d];
    float den = a * a + c * c;
    float inv_den = 1.0f / den;
    float ee = __expf(-d2 * inv_den);
    float loc = sqrtf(2.0f * a * c * inv_den);
    float cc = __cosf(tq[d] - tk[d]);
    sum += eq[d] * ek[d] * loc * ee * cc;
  }
  out[idx] = sum * 3.35462627902511838821e-4f; // * exp(-8)
}

// ---- fused attention: one wave per (b,h,16-query tile); S in LDS; packed K/V ----
__global__ __launch_bounds__(32) void attn_kernel(
    const __bf16* __restrict__ qb, const __bf16* __restrict__ kpack,
    const __bf16* __restrict__ vpack, const float* __restrict__ kern_s,
    __bf16* __restrict__ ctxb) {
  __shared__ float S[16 * LL]; // 32 KB of 320 KB/WGP
  int lane = threadIdx.x;
  int qt = blockIdx.x, h = blockIdx.y, b = blockIdx.z;
  int half = lane >> 4, nn = lane & 15;
  const __bf16* Q  = qb + (size_t)b * LL * DMODEL + h * DH;
  const __bf16* Kp = kpack + (size_t)(b * HHEADS + h) * 2 * 32 * 512;
  const __bf16* Vp = vpack + (size_t)(b * HHEADS + h) * 16 * 4 * 512;

  v16bf a0 = frag_a(Q, DMODEL, qt * 16, 0, lane);
  v16bf a1 = frag_a(Q, DMODEL, qt * 16, 32, lane);
  for (int kt = 0; kt < LL / 16; ++kt) {
    v8f acc = {};
    v16bf b0 = *(const v16bf*)(Kp + ((size_t)(0 * 32 + kt)) * 512 + lane * 16);
    v16bf b1 = *(const v16bf*)(Kp + ((size_t)(1 * 32 + kt)) * 512 + lane * 16);
    acc = __builtin_amdgcn_wmma_f32_16x16x32_bf16(false, a0, false, b0, (short)0, acc, false, false);
    acc = __builtin_amdgcn_wmma_f32_16x16x32_bf16(false, a1, false, b1, (short)0, acc, false, false);
#pragma unroll
    for (int i = 0; i < 8; ++i) {
      int m = i + half * 8;
      int kc = kt * 16 + nn;
      S[m * LL + kc] = acc[i] * 0.125f + kern_s[((size_t)b * LL + qt * 16 + m) * LL + kc];
    }
  }
  __syncthreads();

  // row softmax over 512 keys
  for (int r = 0; r < 16; ++r) {
    float mx = -3.4e38f;
    for (int c = lane; c < LL; c += 32) mx = fmaxf(mx, S[r * LL + c]);
#pragma unroll
    for (int o = 16; o; o >>= 1) mx = fmaxf(mx, __shfl_xor(mx, o, 32));
    float sm = 0.0f;
    for (int c = lane; c < LL; c += 32) {
      float e = __expf(S[r * LL + c] - mx);
      S[r * LL + c] = e;
      sm += e;
    }
#pragma unroll
    for (int o = 16; o; o >>= 1) sm += __shfl_xor(sm, o, 32);
    float inv = 1.0f / sm;
    for (int c = lane; c < LL; c += 32) S[r * LL + c] *= inv;
  }
  __syncthreads();

  // ctx = P(16x512) x V(512x64)
  int r = lane & 15;
#pragma unroll
  for (int nt = 0; nt < 4; ++nt) {
    v8f acc = {};
    for (int k0 = 0; k0 < LL; k0 += 32) {
      v16bf ap;
#pragma unroll
      for (int e = 0; e < 16; ++e) ap[e] = (__bf16)S[r * LL + k0 + kmap(e, half)];
      v16bf bp = *(const v16bf*)(Vp + (((size_t)(k0 >> 5)) * 4 + nt) * 512 + lane * 16);
      acc = __builtin_amdgcn_wmma_f32_16x16x32_bf16(false, ap, false, bp, (short)0, acc, false, false);
    }
#pragma unroll
    for (int i = 0; i < 8; ++i) {
      int m = i + half * 8;
      ctxb[((size_t)b * LL + qt * 16 + m) * DMODEL + h * DH + nt * 16 + nn] = (__bf16)acc[i];
    }
  }
}

// ---- residual add + LayerNorm (one block per row of 512) ----
__global__ __launch_bounds__(256) void add_ln_kernel(
    const float* x, const float* r,
    const float* __restrict__ g, const float* __restrict__ bta,
    float* xout, __bf16* xbout) {
  __shared__ float red[256];
  int row = blockIdx.x, tid = threadIdx.x;
  size_t basei = (size_t)row * DMODEL;
  float v0 = x[basei + tid] + r[basei + tid];
  float v1 = x[basei + tid + 256] + r[basei + tid + 256];
  red[tid] = v0 + v1;
  __syncthreads();
  for (int s = 128; s; s >>= 1) {
    if (tid < s) red[tid] += red[tid + s];
    __syncthreads();
  }
  float mean = red[0] * (1.0f / DMODEL);
  __syncthreads();
  float d0 = v0 - mean, d1 = v1 - mean;
  red[tid] = d0 * d0 + d1 * d1;
  __syncthreads();
  for (int s = 128; s; s >>= 1) {
    if (tid < s) red[tid] += red[tid + s];
    __syncthreads();
  }
  float inv = rsqrtf(red[0] * (1.0f / DMODEL) + 1e-6f);
  float y0 = d0 * inv * g[tid] + bta[tid];
  float y1 = d1 * inv * g[tid + 256] + bta[tid + 256];
  xout[basei + tid] = y0;
  xout[basei + tid + 256] = y1;
  if (xbout) {
    xbout[basei + tid] = (__bf16)y0;
    xbout[basei + tid + 256] = (__bf16)y1;
  }
}

extern "C" void kernel_launch(void* const* d_in, const int* in_sizes, int n_in,
                              void* d_out, int out_size, void* d_ws, size_t ws_size,
                              hipStream_t stream) {
  (void)in_sizes; (void)n_in; (void)out_size; (void)ws_size;
  const float* x_in = (const float*)d_in[0];
  const float* t    = (const float*)d_in[1];
  const float* wq = (const float*)d_in[2];  const float* bq = (const float*)d_in[3];
  const float* wk = (const float*)d_in[4];  const float* bk = (const float*)d_in[5];
  const float* wv = (const float*)d_in[6];  const float* bv = (const float*)d_in[7];
  const float* wo = (const float*)d_in[8];  const float* bo = (const float*)d_in[9];
  const float* w1 = (const float*)d_in[10]; const float* b1 = (const float*)d_in[11];
  const float* w2 = (const float*)d_in[12]; const float* b2 = (const float*)d_in[13];
  const float* ln1g = (const float*)d_in[14]; const float* ln1b = (const float*)d_in[15];
  const float* ln2g = (const float*)d_in[16]; const float* ln2b = (const float*)d_in[17];
  const float* pw1 = (const float*)d_in[18]; const float* pb1 = (const float*)d_in[19];
  const float* pw2 = (const float*)d_in[20]; const float* pb2 = (const float*)d_in[21];
  const float* sw1 = (const float*)d_in[22]; const float* sb1 = (const float*)d_in[23];
  const float* sw2 = (const float*)d_in[24]; const float* sb2 = (const float*)d_in[25];
  const float* bw1 = (const float*)d_in[26]; const float* bb1 = (const float*)d_in[27];
  const float* bw2 = (const float*)d_in[28]; const float* bb2 = (const float*)d_in[29];

  char* base = (char*)d_ws;
  size_t off = 0;
  auto alloc = [&](size_t bytes) -> void* {
    void* p = base + off;
    off += (bytes + 255) & ~(size_t)255;
    return p;
  };

  float*  xf    = (float*) alloc((size_t)MROWS * DMODEL * 4);
  float*  tmpf  = (float*) alloc((size_t)MROWS * DMODEL * 4);
  __bf16* xb    = (__bf16*)alloc((size_t)MROWS * DMODEL * 2);
  __bf16* qb    = (__bf16*)alloc((size_t)MROWS * DMODEL * 2);
  __bf16* kbuf  = (__bf16*)alloc((size_t)MROWS * DMODEL * 2);
  __bf16* vbuf  = (__bf16*)alloc((size_t)MROWS * DMODEL * 2);
  __bf16* kpk   = (__bf16*)alloc((size_t)MROWS * DMODEL * 2);
  __bf16* vpk   = (__bf16*)alloc((size_t)MROWS * DMODEL * 2);
  __bf16* ctxb  = (__bf16*)alloc((size_t)MROWS * DMODEL * 2);
  __bf16* hb    = (__bf16*)alloc((size_t)MROWS * DFF * 2);
  float*  kerns = (float*) alloc((size_t)BB * LL * LL * 4);
  float*  tpb   = (float*) alloc((size_t)BB * LL * TD * 4);
  float*  sgb   = (float*) alloc((size_t)BB * LL * TD * 4);
  float*  beb   = (float*) alloc((size_t)BB * LL * TD * 4);
  __bf16* wqp   = (__bf16*)alloc((size_t)NL * DMODEL * DMODEL * 2);
  __bf16* wkp   = (__bf16*)alloc((size_t)NL * DMODEL * DMODEL * 2);
  __bf16* wvp   = (__bf16*)alloc((size_t)NL * DMODEL * DMODEL * 2);
  __bf16* wop   = (__bf16*)alloc((size_t)NL * DMODEL * DMODEL * 2);
  __bf16* w1p   = (__bf16*)alloc((size_t)NL * DMODEL * DFF * 2);
  __bf16* w2p   = (__bf16*)alloc((size_t)NL * DFF * DMODEL * 2);

  // pack all weights into WMMA fragment layout (once per launch)
  auto packw = [&](const float* W, __bf16* P, int K, int N) {
    int n = K * N;
    pack_w_kernel<<<dim3((n + 255) / 256), dim3(256), 0, stream>>>(W, P, K, N);
  };
  for (int i = 0; i < NL; ++i) {
    packw(wq + (size_t)i * DMODEL * DMODEL, wqp + (size_t)i * DMODEL * DMODEL, DMODEL, DMODEL);
    packw(wk + (size_t)i * DMODEL * DMODEL, wkp + (size_t)i * DMODEL * DMODEL, DMODEL, DMODEL);
    packw(wv + (size_t)i * DMODEL * DMODEL, wvp + (size_t)i * DMODEL * DMODEL, DMODEL, DMODEL);
    packw(wo + (size_t)i * DMODEL * DMODEL, wop + (size_t)i * DMODEL * DMODEL, DMODEL, DMODEL);
    packw(w1 + (size_t)i * DMODEL * DFF,    w1p + (size_t)i * DMODEL * DFF,    DMODEL, DFF);
    packw(w2 + (size_t)i * DFF * DMODEL,    w2p + (size_t)i * DFF * DMODEL,    DFF, DMODEL);
  }
  f2bf_kernel<<<dim3((MROWS * DMODEL) / 256), dim3(256), 0, stream>>>(x_in, xb, MROWS * DMODEL);

  const float* xcur = x_in;
  for (int i = 0; i < NL; ++i) {
    time_ffn_kernel<<<dim3(MROWS / 256), dim3(256), 0, stream>>>(
        t, pw1 + i * TD, pb1 + i * TD, pw2 + i * TD * TD, pb2 + i * TD,
        sw1 + i * TD, sb1 + i * TD, sw2 + i * TD * TD, sb2 + i * TD,
        bw1 + i * TD, bb1 + i * TD, bw2 + i * TD * TD, bb2 + i * TD,
        tpb, sgb, beb);
    kern_kernel<<<dim3((BB * LL * LL) / 256), dim3(256), 0, stream>>>(t, tpb, sgb, beb, kerns);

    const __bf16* Wq = wqp + (size_t)i * DMODEL * DMODEL;
    const __bf16* Wk = wkp + (size_t)i * DMODEL * DMODEL;
    const __bf16* Wv = wvp + (size_t)i * DMODEL * DMODEL;
    const __bf16* Wo = wop + (size_t)i * DMODEL * DMODEL;
    const __bf16* W1 = w1p + (size_t)i * DMODEL * DFF;
    const __bf16* W2 = w2p + (size_t)i * DFF * DMODEL;

    dim3 gD(MROWS / 16, DMODEL / 64);
    gemm_bf16<<<gD, 128, 0, stream>>>(xb, Wq, bq + i * DMODEL, nullptr, qb, MROWS, DMODEL, DMODEL, 0);
    gemm_bf16<<<gD, 128, 0, stream>>>(xb, Wk, bk + i * DMODEL, nullptr, kbuf, MROWS, DMODEL, DMODEL, 0);
    gemm_bf16<<<gD, 128, 0, stream>>>(xb, Wv, bv + i * DMODEL, nullptr, vbuf, MROWS, DMODEL, DMODEL, 0);

    pack_k_kernel<<<dim3((MROWS * DMODEL) / 256), dim3(256), 0, stream>>>(kbuf, kpk);
    pack_v_kernel<<<dim3((MROWS * DMODEL) / 256), dim3(256), 0, stream>>>(vbuf, vpk);

    attn_kernel<<<dim3(LL / 16, HHEADS, BB), 32, 0, stream>>>(qb, kpk, vpk, kerns, ctxb);

    gemm_bf16<<<gD, 128, 0, stream>>>(ctxb, Wo, bo + i * DMODEL, tmpf, nullptr, MROWS, DMODEL, DMODEL, 0);
    add_ln_kernel<<<dim3(MROWS), dim3(256), 0, stream>>>(xcur, tmpf, ln1g + i * DMODEL, ln1b + i * DMODEL, xf, xb);

    gemm_bf16<<<dim3(MROWS / 16, DFF / 64), 128, 0, stream>>>(xb, W1, b1 + i * DFF, nullptr, hb, MROWS, DFF, DMODEL, 1);
    gemm_bf16<<<gD, 128, 0, stream>>>(hb, W2, b2 + i * DMODEL, tmpf, nullptr, MROWS, DMODEL, DFF, 0);

    float* outx = (i == NL - 1) ? (float*)d_out : xf;
    __bf16* outb = (i == NL - 1) ? nullptr : xb;
    add_ln_kernel<<<dim3(MROWS), dim3(256), 0, stream>>>(xf, tmpf, ln2g + i * DMODEL, ln2b + i * DMODEL, outx, outb);
    xcur = xf;
  }
}
